// GPT_81999515615550
// MI455X (gfx1250) — matrix operand order repaired and checked by reading, not compiled, not run on hardware
//
#include <hip/hip_runtime.h>
#include <hip/hip_fp16.h>
#include <math.h>

typedef __attribute__((ext_vector_type(16))) _Float16 v16h;
typedef __attribute__((ext_vector_type(8)))  _Float16 v8h;
typedef __attribute__((ext_vector_type(8)))  float    v8f;

#define BATCH 1024
#define TSEQ  64
#define CEMB  128
#define NHEAD 8
#define HDIM  16
#define DFF   512
#define VOCAB 65
#define ROWS  (BATCH*TSEQ)   // 65536

// ---------------- embedding: x = tok_emb[inp] + pos_emb ----------------
__global__ void embed_k(const int* __restrict__ inp, const float* __restrict__ tok,
                        const float* __restrict__ pos, float* __restrict__ x) {
    int tid = blockIdx.x * blockDim.x + threadIdx.x;
    if (tid >= ROWS * CEMB) return;
    int r = tid / CEMB, c = tid % CEMB;
    x[tid] = tok[inp[r] * CEMB + c] + pos[(r % TSEQ) * CEMB + c];
}

// ---------------- layernorm over 128 cols, f16 output ----------------
__global__ void ln_k(const float* __restrict__ x, const float* __restrict__ g,
                     const float* __restrict__ b, _Float16* __restrict__ out, int ldo) {
    __shared__ float red[CEMB];
    int r = blockIdx.x, c = threadIdx.x;
    float v = x[(long long)r * CEMB + c];
    red[c] = v; __syncthreads();
    for (int s = 64; s > 0; s >>= 1) { if (c < s) red[c] += red[c + s]; __syncthreads(); }
    float mu = red[0] * (1.0f / CEMB); __syncthreads();
    float d = v - mu;
    red[c] = d * d; __syncthreads();
    for (int s = 64; s > 0; s >>= 1) { if (c < s) red[c] += red[c + s]; __syncthreads(); }
    float var = red[0] * (1.0f / CEMB);
    float y = d * rsqrtf(var + 1e-5f) * g[c] + b[c];
    out[(long long)r * ldo + c] = (_Float16)y;
}

// ---------------- weight prep: w[K,N] f32 -> Bt[Npad,K] f16 (zero pad) -----------
__global__ void wprep_k(const float* __restrict__ w, _Float16* __restrict__ bt,
                        int K, int N, int Npad) {
    int tid = blockIdx.x * blockDim.x + threadIdx.x;
    if (tid >= Npad * K) return;
    int n = tid / K, k = tid % K;
    bt[tid] = (n < N) ? (_Float16)w[k * N + n] : (_Float16)0.0f;
}

// ---------------- WMMA GEMM: 16x(16*NT) output strip per wave --------------------
// C[M,N] = A[M,K](f16) * Bt[N,K]^T(f16); K is compile-time (128 or 512) so the
// k-loop fully unrolls: no loop SALU, all addressing folded into load offsets.
// modes: 0 plain f32, 1 += Cf (residual in place), 2 GELU -> f16 (Ch), 3 +bias f32
template<int NT, int K>
__global__ __launch_bounds__(256) void gemm_k(
    const _Float16* __restrict__ A, int lda,
    const _Float16* __restrict__ Bt,
    float* __restrict__ Cf, _Float16* __restrict__ Ch, int ldc,
    int Mtiles, int Ngroups, int Nact, int mode, const float* __restrict__ bias) {
    int wave = threadIdx.x >> 5;
    int lane = threadIdx.x & 31;
    int wt = blockIdx.x * 8 + wave;
    if (wt >= Mtiles * Ngroups) return;           // uniform per wave: EXEC stays full
    int tm = wt / Ngroups, tg = wt % Ngroups;
    int row0 = tm * 16, col0 = tg * (16 * NT);
    int lh = lane & 15, hi = lane >> 4;

    const _Float16* arow = A  + (long long)(row0 + lh) * lda;
    const _Float16* brow = Bt + (long long)(col0 + lh) * K;

    __builtin_prefetch(arow, 0, 0);               // global_prefetch_b8, outside hot loop

    v8f zero = {};
    v8f acc[NT];
    #pragma unroll
    for (int j = 0; j < NT; j++) acc[j] = zero;

    #pragma unroll
    for (int kk = 0; kk < K; kk += 32) {
        // A 16x32 f16: lane<16 -> K {kk..kk+7, kk+16..kk+23}; lane>=16 shifted by 8
        v8h a0 = *(const v8h*)(arow + kk + hi * 8);
        v8h a1 = *(const v8h*)(arow + kk + 16 + hi * 8);
        union { v8h h[2]; v16h v; } ua;
        ua.h[0] = a0; ua.h[1] = a1;
        #pragma unroll
        for (int j = 0; j < NT; j++) {
            const _Float16* bj = brow + (long long)j * 16 * K;
            // B 32x16 f16: lane<16 -> K {kk..kk+7, kk+16..kk+23}; lane>=16 shifted by 8
            v8h b0 = *(const v8h*)(bj + kk + hi * 8);
            v8h b1 = *(const v8h*)(bj + kk + 16 + hi * 8);
            union { v8h h[2]; v16h v; } ub;
            ub.h[0] = b0; ub.h[1] = b1;
            acc[j] = __builtin_amdgcn_wmma_f32_16x16x32_f16(
                false, ua.v, false, ub.v, (short)0, acc[j], false, false);
        }
    }

    #pragma unroll
    for (int j = 0; j < NT; j++) {
        int col = col0 + j * 16 + lh;
        if (col < Nact) {
            #pragma unroll
            for (int r = 0; r < 8; r++) {
                int row = row0 + hi * 8 + r;      // C/D layout: VGPR r -> M=r (lo) / M=8+r (hi)
                float v = acc[j][r];
                long long idx = (long long)row * ldc + col;
                if (mode == 1) {
                    v += Cf[idx];
                } else if (mode == 2) {
                    v = 0.5f * v * (1.0f + erff(v * 0.70710678f));
                    Ch[idx] = (_Float16)v;
                    continue;
                } else if (mode == 3) {
                    v += bias[col];
                }
                Cf[idx] = v;
            }
        }
    }
}

// ---------------- attention: one (batch,head) per 64-thread block ----------------
__global__ __launch_bounds__(64) void attn_k(const float* __restrict__ qkv,
                                             _Float16* __restrict__ out) {
    __shared__ float kb[TSEQ][HDIM];
    __shared__ float vb[TSEQ][HDIM];
    int bh = blockIdx.x;
    int b = bh / NHEAD, h = bh % NHEAD;
    int t = threadIdx.x;
    long long rbase = (long long)(b * TSEQ + t) * 384;
    int off = h * HDIM;
    float q[HDIM];
    #pragma unroll
    for (int d = 0; d < HDIM; d++) {
        kb[t][d] = qkv[rbase + 128 + off + d];
        vb[t][d] = qkv[rbase + 256 + off + d];
        q[d]     = qkv[rbase + off + d];
    }
    __syncthreads();
    float sc[TSEQ];
    float m = -1e30f;
    for (int j = 0; j <= t; j++) {
        float s = 0.0f;
        #pragma unroll
        for (int d = 0; d < HDIM; d++) s += q[d] * kb[j][d];
        s *= 0.25f;                               // 1/sqrt(16)
        sc[j] = s;
        if (s > m) m = s;
    }
    float den = 0.0f;
    for (int j = 0; j <= t; j++) { float e = expf(sc[j] - m); sc[j] = e; den += e; }
    float inv = 1.0f / den;
    long long obase = (long long)(b * TSEQ + t) * CEMB + off;
    #pragma unroll
    for (int d = 0; d < HDIM; d++) {
        float o = 0.0f;
        for (int j = 0; j <= t; j++) o += sc[j] * vb[j][d];
        out[obase + d] = (_Float16)(o * inv);
    }
}

// ---------------- cross entropy ----------------
__global__ void zero_k(float* a) { a[0] = 0.0f; a[1] = 0.0f; }

__global__ void loss_k(const float* __restrict__ logits, const int* __restrict__ labels,
                       float* __restrict__ accum) {
    int r = blockIdx.x * blockDim.x + threadIdx.x;
    if (r >= ROWS) return;
    const float* lr = logits + (long long)r * VOCAB;
    float m = -1e30f;
    for (int i = 0; i < VOCAB; i++) m = fmaxf(m, lr[i]);
    float s = 0.0f;
    for (int i = 0; i < VOCAB; i++) s += expf(lr[i] - m);
    float lse = m + logf(s);
    int lab = labels[r];
    if (lab != -1) {
        atomicAdd(&accum[0], lse - lr[lab]);
        atomicAdd(&accum[1], 1.0f);
    }
}

__global__ void fin_k(const float* __restrict__ accum, float* __restrict__ out) {
    out[0] = accum[0] / fmaxf(accum[1], 1.0f);
}

// =========================== host driver ===========================
extern "C" void kernel_launch(void* const* d_in, const int* in_sizes, int n_in,
                              void* d_out, int out_size, void* d_ws, size_t ws_size,
                              hipStream_t stream) {
    const int*   inp    = (const int*)  d_in[0];
    const int*   labels = (const int*)  d_in[1];
    const float* tok    = (const float*)d_in[2];
    const float* pos    = (const float*)d_in[3];
    const float* ln1g   = (const float*)d_in[4];
    const float* ln1b   = (const float*)d_in[5];
    const float* wqkv   = (const float*)d_in[6];
    const float* wproj  = (const float*)d_in[7];
    const float* ln2g   = (const float*)d_in[8];
    const float* ln2b   = (const float*)d_in[9];
    const float* wfc1   = (const float*)d_in[10];
    const float* wfc2   = (const float*)d_in[11];
    const float* lnfg   = (const float*)d_in[12];
    const float* lnfb   = (const float*)d_in[13];
    const float* wlm    = (const float*)d_in[14];
    const float* blm    = (const float*)d_in[15];
    float* out = (float*)d_out;

    char* ws = (char*)d_ws;
    size_t o = 0;
    float*    x    = (float*)(ws + o);    o += (size_t)ROWS * CEMB * 4;   // 32 MB
    float*    g    = (float*)(ws + o);    o += (size_t)ROWS * 384  * 4;   // 96 MB (qkv f32)
    _Float16* h16  = (_Float16*)(ws + o); o += (size_t)ROWS * CEMB * 2;   // 16 MB (GEMM A)
    _Float16* h16b = (_Float16*)(ws + o); o += (size_t)ROWS * DFF  * 2;   // 64 MB (fc1 out f16)
    _Float16* wt   = (_Float16*)(ws + o); o += (size_t)512 * 512 * 2;
    float*    accum= (float*)(ws + o);

    const int Mt = ROWS / 16;                     // 4096 row tiles

    embed_k<<<(ROWS * CEMB) / 256, 256, 0, stream>>>(inp, tok, pos, x);

    for (int l = 0; l < 6; l++) {
        // --- attention half ---
        ln_k<<<ROWS, CEMB, 0, stream>>>(x, ln1g + l * CEMB, ln1b + l * CEMB, h16, CEMB);
        wprep_k<<<(384 * 128 + 255) / 256, 256, 0, stream>>>(
            wqkv + (size_t)l * CEMB * 384, wt, CEMB, 384, 384);
        gemm_k<4, 128><<<(Mt * 6) / 8, 256, 0, stream>>>(h16, CEMB, wt, g, nullptr, 384,
                                                         Mt, 6, 384, 0, nullptr);
        attn_k<<<BATCH * NHEAD, 64, 0, stream>>>(g, h16);
        wprep_k<<<(128 * 128 + 255) / 256, 256, 0, stream>>>(
            wproj + (size_t)l * CEMB * CEMB, wt, CEMB, CEMB, CEMB);
        gemm_k<4, 128><<<(Mt * 2) / 8, 256, 0, stream>>>(h16, CEMB, wt, x, nullptr, CEMB,
                                                         Mt, 2, CEMB, 1, nullptr);
        // --- MLP half ---
        ln_k<<<ROWS, CEMB, 0, stream>>>(x, ln2g + l * CEMB, ln2b + l * CEMB, h16, CEMB);
        wprep_k<<<(512 * 128 + 255) / 256, 256, 0, stream>>>(
            wfc1 + (size_t)l * CEMB * DFF, wt, CEMB, DFF, DFF);
        gemm_k<4, 128><<<(Mt * 8) / 8, 256, 0, stream>>>(h16, CEMB, wt, nullptr, h16b, DFF,
                                                         Mt, 8, DFF, 2, nullptr);
        wprep_k<<<(128 * 512 + 255) / 256, 256, 0, stream>>>(
            wfc2 + (size_t)l * DFF * CEMB, wt, DFF, CEMB, CEMB);
        gemm_k<4, 512><<<(Mt * 2) / 8, 256, 0, stream>>>(h16b, DFF, wt, x, nullptr, CEMB,
                                                         Mt, 2, CEMB, 1, nullptr);
    }

    // --- final LN + lm head (N=65 padded to 80, one wave covers all 5 N-tiles) ---
    ln_k<<<ROWS, CEMB, 0, stream>>>(x, lnfg, lnfb, h16, CEMB);
    wprep_k<<<(80 * 128 + 255) / 256, 256, 0, stream>>>(wlm, wt, CEMB, VOCAB, 80);
    gemm_k<5, 128><<<Mt / 8, 256, 0, stream>>>(h16, CEMB, wt, out, nullptr, VOCAB,
                                               Mt, 1, VOCAB, 3, blm);

    // --- loss ---
    zero_k<<<1, 1, 0, stream>>>(accum);
    loss_k<<<ROWS / 256, 256, 0, stream>>>(out, labels, accum);
    fin_k<<<1, 1, 0, stream>>>(accum, out + (size_t)ROWS * VOCAB);
}
